// TrellisMoELayer_45887430590501
// MI455X (gfx1250) — matrix operand order repaired and checked by reading, not compiled, not run on hardware
//
#include <hip/hip_runtime.h>
#include <hip/hip_bf16.h>
#include <math.h>

// MoE SwiGLU layer for MI455X (gfx1250), bf16 WMMA pipeline.
// E=16 experts, D=1024, F=512, T=B*S=4096, top_k=8.
//
// v2: LDS-staged B panels via global_load_async_to_lds_b128 (ASYNCcnt,
// double-buffered), 2 M-tiles per wave, 4 WMMAs per staged K-substep.

typedef __bf16 bf16_t;
typedef __attribute__((ext_vector_type(16))) __bf16 v16bf;
typedef __attribute__((ext_vector_type(8)))  __bf16 v8bf;
typedef __attribute__((ext_vector_type(8)))  float  v8f;

#define E_ 16
#define D_ 1024
#define F_ 512
#define T_ 4096

// LDS row padding: 64 data elems + 8 pad = 72 elems (144 B = 36 dwords).
// 16 rows * 36 dwords mod 64 banks -> 16 distinct multiples of 4: conflict-free
// quad-bank spans for per-lane b128 reads.
#define KP_   64
#define LROW_ 72

// -------------------------------------------------------------------------
// CDNA5 async global->LDS copy (ASYNCcnt path). The VDST operand is the
// 32-bit LDS byte offset; generic LDS pointers carry it in their low 32 bits.
// -------------------------------------------------------------------------
__device__ __forceinline__ void async_b128_to_lds(void* lds_ptr, const void* gptr)
{
    unsigned lds_off = (unsigned)(size_t)lds_ptr;
    unsigned long long ga = (unsigned long long)(size_t)gptr;
    asm volatile("global_load_async_to_lds_b128 %0, %1, off"
                 :: "v"(lds_off), "v"(ga) : "memory");
}

__device__ __forceinline__ void wait_asynccnt0()
{
    asm volatile("s_wait_asynccnt 0x0" ::: "memory");
}

// -------------------------------------------------------------------------
// Router: one wave per token. logits -> softmax -> top-k -> renormalize.
// -------------------------------------------------------------------------
__global__ void moe_router_kernel(const float* __restrict__ x,
                                  const float* __restrict__ rw,
                                  const int* __restrict__ topk,
                                  float* __restrict__ cw)
{
    const int wave = threadIdx.x >> 5;
    const int lane = threadIdx.x & 31;
    const int t = blockIdx.x * (blockDim.x >> 5) + wave;
    if (t >= T_) return;

    float logit[E_];
    const float* xr = x + (size_t)t * D_;
    for (int e = 0; e < E_; ++e) {
        const float* wr = rw + (size_t)e * D_;
        float p = 0.f;
        for (int d = lane; d < D_; d += 32) p += xr[d] * wr[d];
        for (int off = 16; off > 0; off >>= 1) p += __shfl_xor(p, off, 32);
        logit[e] = p;            // all lanes hold the full sum
    }
    float mx = logit[0];
    for (int e = 1; e < E_; ++e) mx = fmaxf(mx, logit[e]);
    float pr[E_];
    float s = 0.f;
    for (int e = 0; e < E_; ++e) { pr[e] = __expf(logit[e] - mx); s += pr[e]; }
    for (int e = 0; e < E_; ++e) pr[e] /= s;

    int K = *topk; if (K > E_) K = E_; if (K < 1) K = 1;
    bool sel[E_];
    for (int e = 0; e < E_; ++e) sel[e] = false;
    float tsum = 0.f;
    for (int i = 0; i < K; ++i) {           // deterministic argmax scan
        int bj = 0; float bv = -1.f;
        for (int e = 0; e < E_; ++e)
            if (!sel[e] && pr[e] > bv) { bv = pr[e]; bj = e; }
        sel[bj] = true; tsum += bv;
    }
    if (lane < E_)
        cw[(size_t)t * E_ + lane] = sel[lane] ? (pr[lane] / tsum) : 0.f;
}

// -------------------------------------------------------------------------
// Elementwise fp32 -> bf16
// -------------------------------------------------------------------------
__global__ void f32_to_bf16_kernel(const float* __restrict__ src,
                                   bf16_t* __restrict__ dst, int n)
{
    int i = blockIdx.x * blockDim.x + threadIdx.x;
    if (i < n) dst[i] = (bf16_t)src[i];
}

// -------------------------------------------------------------------------
// LDS-tiled transpose: src fp32 [R,C] -> dst bf16 [C,R]; batched via blockIdx.z
// -------------------------------------------------------------------------
__global__ void transpose_f32_to_bf16(const float* __restrict__ src,
                                      bf16_t* __restrict__ dst, int R, int C)
{
    __shared__ float tile[32][33];
    const float* s = src + (size_t)blockIdx.z * R * C;
    bf16_t*      d = dst + (size_t)blockIdx.z * R * C;
    const int x  = blockIdx.x * 32 + threadIdx.x;   // src col
    const int y0 = blockIdx.y * 32;                 // src row base
    for (int i = threadIdx.y; i < 32; i += 8) {
        int y = y0 + i;
        if (y < R && x < C) tile[i][threadIdx.x] = s[(size_t)y * C + x];
    }
    __syncthreads();
    for (int i = threadIdx.y; i < 32; i += 8) {
        int r = blockIdx.x * 32 + i;        // dst row (= src col)
        int c = y0 + threadIdx.x;           // dst col (= src row)
        if (r < C && c < R) d[(size_t)r * R + c] = (bf16_t)tile[threadIdx.x][i];
    }
}

// -------------------------------------------------------------------------
// Operand fragment helpers (v_wmma_f32_16x16x32_bf16 register layouts)
// -------------------------------------------------------------------------
union AFrag { v16bf v; v8bf h[2]; };

// A-tile (16x32): lane<16 rows M=lane, K 0..15; lane>=16 K 16..31
__device__ __forceinline__ v16bf load_a_tile(const bf16_t* base, int lane,
                                             int row_stride)
{
    AFrag a;
    const bf16_t* ap = base;   // caller pre-offsets row & lane-K
    (void)row_stride; (void)lane;
    a.h[0] = *reinterpret_cast<const v8bf*>(ap);
    a.h[1] = *reinterpret_cast<const v8bf*>(ap + 8);
    return a.v;
}

// B-tile (32x16) from an LDS panel row: lane holds column n, 16 contiguous K.
__device__ __forceinline__ v16bf load_b_lds(const bf16_t* prow)
{
    AFrag b;
    b.h[0] = *reinterpret_cast<const v8bf*>(prow);
    b.h[1] = *reinterpret_cast<const v8bf*>(prow + 8);
    return b.v;
}

// -------------------------------------------------------------------------
// Fused SwiGLU GEMM: H[T,F] = silu(X @ Wg) * (X @ Wu)
//   X   : [T,D] bf16 row-major
//   WgT, WuT : [F,D] bf16 row-major (pre-transposed)
// Block = 128 threads (4 waves). Block tile: 128 M x 16 N.
// Each wave: 2 M-tiles. B panels (both weights, K=64) staged in LDS by
// async-to-LDS, double buffered.
// -------------------------------------------------------------------------
__global__ void swiglu_gemm_kernel(const bf16_t* __restrict__ X,
                                   const bf16_t* __restrict__ WgT,
                                   const bf16_t* __restrict__ WuT,
                                   bf16_t* __restrict__ H)
{
    __shared__ __align__(16) bf16_t Bp[2][2][16][LROW_];  // [buf][w][n][k]

    const int tid  = threadIdx.x;
    const int wave = tid >> 5;
    const int lane = tid & 31;
    const int n0 = blockIdx.x * 16;
    const int m0 = blockIdx.y * 128 + wave * 32;   // two 16-row tiles

    // async loader task: 256 chunks of 16B (2 weights * 16 rows * 8 k-chunks)
    const int c0 = tid;          // chunk ids c0 and c0+128
    auto issue_panel = [&](int buf, int k0) {
        #pragma unroll
        for (int cc = 0; cc < 2; ++cc) {
            int c  = c0 + cc * 128;
            int w  = c >> 7;
            int r  = (c >> 3) & 15;
            int kc = c & 7;
            const bf16_t* gsrc = (w ? WuT : WgT) +
                                 (size_t)(n0 + r) * D_ + k0 + kc * 8;
            async_b128_to_lds(&Bp[buf][w][r][kc * 8], gsrc);
        }
    };

    // A addressing
    const int arow0 = m0 + (lane & 15);
    const int ak    = (lane >> 4) << 4;     // 0 or 16
    // B addressing (LDS)
    const int brow = lane & 15;
    const int bk   = (lane >> 4) << 4;      // 0 or 16

    v8f accg[2] = {v8f{}, v8f{}};
    v8f accu[2] = {v8f{}, v8f{}};

    issue_panel(0, 0);
    int buf = 0;
    for (int k0 = 0; k0 < D_; k0 += KP_, buf ^= 1) {
        wait_asynccnt0();
        __syncthreads();
        if (k0 + KP_ < D_) issue_panel(buf ^ 1, k0 + KP_);

        #pragma unroll
        for (int ks = 0; ks < KP_; ks += 32) {
            v16bf bg = load_b_lds(&Bp[buf][0][brow][ks + bk]);
            v16bf bu = load_b_lds(&Bp[buf][1][brow][ks + bk]);
            #pragma unroll
            for (int t = 0; t < 2; ++t) {
                const bf16_t* ap = X + (size_t)(arow0 + t * 16) * D_ +
                                   (k0 + ks) + ak;
                v16bf a = load_a_tile(ap, lane, D_);
                accg[t] = __builtin_amdgcn_wmma_f32_16x16x32_bf16(
                    false, a, false, bg, (short)0, accg[t], false, false);
                accu[t] = __builtin_amdgcn_wmma_f32_16x16x32_bf16(
                    false, a, false, bu, (short)0, accu[t], false, false);
            }
        }
    }

    // C layout: VGPR r -> row tile_m0 + r + (lane>=16 ? 8 : 0), col n0+lane%16
    const int col = n0 + (lane & 15);
    #pragma unroll
    for (int t = 0; t < 2; ++t) {
        const int rbase = m0 + t * 16 + ((lane >> 4) << 3);
        #pragma unroll
        for (int r = 0; r < 8; ++r) {
            float g = accg[t][r], u = accu[t][r];
            float sg = g / (1.f + __expf(-g));           // silu
            H[(size_t)(rbase + r) * F_ + col] = (bf16_t)(sg * u);
        }
    }
}

// -------------------------------------------------------------------------
// Down-projection GEMM: OUT[T,D] (=|+=) scale * (H @ Wd)
//   H : [T,F] bf16, WdT : [D,F] bf16 (pre-transposed), OUT fp32
//   scale = cw[t, e] if cw != null else 1.0
// Block = 128 threads; block tile 128 M x 16 N; K panel = 64 staged in LDS.
// -------------------------------------------------------------------------
__global__ void down_gemm_kernel(const bf16_t* __restrict__ Hm,
                                 const bf16_t* __restrict__ WdT,
                                 const float* __restrict__ cw,
                                 int e, int accumulate,
                                 float* __restrict__ OUT)
{
    __shared__ __align__(16) bf16_t Bp[2][16][LROW_];    // [buf][n][k]

    const int tid  = threadIdx.x;
    const int wave = tid >> 5;
    const int lane = tid & 31;
    const int n0 = blockIdx.x * 16;
    const int m0 = blockIdx.y * 128 + wave * 32;

    // 128 chunks (16 rows * 8 k-chunks), one per thread
    auto issue_panel = [&](int buf, int k0) {
        int r  = (tid >> 3) & 15;
        int kc = tid & 7;
        const bf16_t* gsrc = WdT + (size_t)(n0 + r) * F_ + k0 + kc * 8;
        async_b128_to_lds(&Bp[buf][r][kc * 8], gsrc);
    };

    const int arow0 = m0 + (lane & 15);
    const int ak    = (lane >> 4) << 4;
    const int brow  = lane & 15;
    const int bk    = (lane >> 4) << 4;

    v8f acc[2] = {v8f{}, v8f{}};

    issue_panel(0, 0);
    int buf = 0;
    for (int k0 = 0; k0 < F_; k0 += KP_, buf ^= 1) {
        wait_asynccnt0();
        __syncthreads();
        if (k0 + KP_ < F_) issue_panel(buf ^ 1, k0 + KP_);

        #pragma unroll
        for (int ks = 0; ks < KP_; ks += 32) {
            v16bf b = load_b_lds(&Bp[buf][brow][ks + bk]);
            #pragma unroll
            for (int t = 0; t < 2; ++t) {
                const bf16_t* ap = Hm + (size_t)(arow0 + t * 16) * F_ +
                                   (k0 + ks) + ak;
                v16bf a = load_a_tile(ap, lane, F_);
                acc[t] = __builtin_amdgcn_wmma_f32_16x16x32_bf16(
                    false, a, false, b, (short)0, acc[t], false, false);
            }
        }
    }

    const int col = n0 + (lane & 15);
    #pragma unroll
    for (int t = 0; t < 2; ++t) {
        const int rbase = m0 + t * 16 + ((lane >> 4) << 3);
        #pragma unroll
        for (int r = 0; r < 8; ++r) {
            const int row = rbase + r;
            float v = acc[t][r];
            if (cw) v *= cw[(size_t)row * E_ + e];
            if (accumulate) OUT[(size_t)row * D_ + col] += v;
            else            OUT[(size_t)row * D_ + col]  = v;
        }
    }
}

// -------------------------------------------------------------------------
// Host launcher
// -------------------------------------------------------------------------
extern "C" void kernel_launch(void* const* d_in, const int* in_sizes, int n_in,
                              void* d_out, int out_size, void* d_ws, size_t ws_size,
                              hipStream_t stream)
{
    (void)in_sizes; (void)n_in; (void)out_size; (void)ws_size;
    const float* x  = (const float*)d_in[0];
    const float* rw = (const float*)d_in[1];
    const float* gw = (const float*)d_in[2];   // [E,D,F]
    const float* uw = (const float*)d_in[3];   // [E,D,F]
    const float* dw = (const float*)d_in[4];   // [E,F,D]
    const float* sg = (const float*)d_in[5];   // [D,F]
    const float* su = (const float*)d_in[6];   // [D,F]
    const float* sd = (const float*)d_in[7];   // [F,D]
    const int*   tk = (const int*)d_in[8];
    float* out = (float*)d_out;

    char* ws = (char*)d_ws;
    size_t off = 0;
    auto wsalloc = [&](size_t bytes) -> void* {
        void* p = ws + off;
        off += (bytes + 255) & ~(size_t)255;
        return p;
    };
    float*  cw   = (float*) wsalloc((size_t)T_ * E_ * sizeof(float));
    bf16_t* xb   = (bf16_t*)wsalloc((size_t)T_ * D_ * sizeof(bf16_t));
    bf16_t* gT   = (bf16_t*)wsalloc((size_t)E_ * F_ * D_ * sizeof(bf16_t));
    bf16_t* uT   = (bf16_t*)wsalloc((size_t)E_ * F_ * D_ * sizeof(bf16_t));
    bf16_t* dT   = (bf16_t*)wsalloc((size_t)E_ * D_ * F_ * sizeof(bf16_t));
    bf16_t* sgT  = (bf16_t*)wsalloc((size_t)F_ * D_ * sizeof(bf16_t));
    bf16_t* suT  = (bf16_t*)wsalloc((size_t)F_ * D_ * sizeof(bf16_t));
    bf16_t* sdT  = (bf16_t*)wsalloc((size_t)D_ * F_ * sizeof(bf16_t));
    bf16_t* hbuf = (bf16_t*)wsalloc((size_t)T_ * F_ * sizeof(bf16_t));

    // Router combine weights
    moe_router_kernel<<<T_ / 4, 128, 0, stream>>>(x, rw, tk, cw);

    // Stage activations + transposed weights as bf16
    f32_to_bf16_kernel<<<(T_ * D_) / 256, 256, 0, stream>>>(x, xb, T_ * D_);
    transpose_f32_to_bf16<<<dim3(F_/32, D_/32, E_), dim3(32, 8), 0, stream>>>(gw, gT, D_, F_);
    transpose_f32_to_bf16<<<dim3(F_/32, D_/32, E_), dim3(32, 8), 0, stream>>>(uw, uT, D_, F_);
    transpose_f32_to_bf16<<<dim3(D_/32, F_/32, E_), dim3(32, 8), 0, stream>>>(dw, dT, F_, D_);
    transpose_f32_to_bf16<<<dim3(F_/32, D_/32, 1 ), dim3(32, 8), 0, stream>>>(sg, sgT, D_, F_);
    transpose_f32_to_bf16<<<dim3(F_/32, D_/32, 1 ), dim3(32, 8), 0, stream>>>(su, suT, D_, F_);
    transpose_f32_to_bf16<<<dim3(D_/32, F_/32, 1 ), dim3(32, 8), 0, stream>>>(sd, sdT, F_, D_);

    // Shared expert first: initializes OUT with '='
    swiglu_gemm_kernel<<<dim3(F_/16, T_/128), 128, 0, stream>>>(xb, sgT, suT, hbuf);
    down_gemm_kernel<<<dim3(D_/16, T_/128), 128, 0, stream>>>(hbuf, sdT, nullptr, 0, 0, out);

    // Experts: weighted accumulation into OUT
    for (int e = 0; e < E_; ++e) {
        swiglu_gemm_kernel<<<dim3(F_/16, T_/128), 128, 0, stream>>>(
            xb, gT + (size_t)e * F_ * D_, uT + (size_t)e * F_ * D_, hbuf);
        down_gemm_kernel<<<dim3(D_/16, T_/128), 128, 0, stream>>>(
            hbuf, dT + (size_t)e * D_ * F_, cw, e, 1, out);
    }
}
